// TemporalFrequencyMasking_25151328485772
// MI455X (gfx1250) — matrix operand order repaired and checked by reading, not compiled, not run on hardware
//
#include <hip/hip_runtime.h>
#include <hip/hip_bf16.h>
#include <math.h>
#include <float.h>

typedef __attribute__((ext_vector_type(16))) __bf16 v16bf;
typedef __attribute__((ext_vector_type(8)))  float  v8f;

namespace {

constexpr int kB  = 2, kT = 512, kC = 263, kD = 512;
constexpr int kKP = 288;              // C padded to multiple of 32 (WMMA K)
constexpr int kM  = kB * kT;          // 1024 GEMM rows
constexpr int kF  = kT / 2 + 1;       // 257 rfft bins
constexpr int kNMT = 51, kNMF = 51;   // int(T*0.1)
constexpr int kNPAD = 272;            // C padded to multiple of 16 (proj N)
constexpr float kPI = 3.14159265358979323846f;

// d_out layout (floats): out_t [B,T,D] | tidx [B,51] | out_f [B,T,C] | fidx [B,51]
constexpr int kOutTI = kM * kD;              // 524288
constexpr int kOutF  = kOutTI + kB * kNMT;   // 524390
constexpr int kOutFI = kOutF + kM * kC;      // 793702

__device__ __forceinline__ unsigned short f2bf(float f) {
  unsigned int u = __float_as_uint(f);
  u += 0x7FFFu + ((u >> 16) & 1u);       // round-to-nearest-even
  return (unsigned short)(u >> 16);
}

union FragU { uint4 q[2]; v16bf v; };

// One wave computes a 16x16 f32 tile of  A[M,K] * B^T  where Bt row n holds
// output-column n over K (both bf16, row-major, 16B-aligned rows).
// A layout (ISA 7.12.2, 16-bit A 16x32): lane r (0..15) row r, half-wave selects
// K sub-octet; B layout mirrors A: half-wave selects K sub-octet.
__device__ __forceinline__ v8f wmma_tile16(const unsigned short* __restrict__ A, int lda,
                                           const unsigned short* __restrict__ Bt, int ldb,
                                           int m0, int n0, int K) {
  const int lane = threadIdx.x & 31;
  const int r    = lane & 15;
  const int hi   = lane >> 4;
  const unsigned short* arow = A  + (size_t)(m0 + r) * lda + hi * 8;
  const unsigned short* brow = Bt + (size_t)(n0 + r) * ldb + hi * 8;
  v8f acc = {};
  for (int k = 0; k < K; k += 32) {
    FragU a, b;
    a.q[0] = *(const uint4*)(arow + k);
    a.q[1] = *(const uint4*)(arow + k + 16);
    b.q[0] = *(const uint4*)(brow + k);
    b.q[1] = *(const uint4*)(brow + k + 16);
    __builtin_prefetch(arow + k + 32, 0, 3);   // global_prefetch_b8
    __builtin_prefetch(brow + k + 32, 0, 3);
    acc = __builtin_amdgcn_wmma_f32_16x16x32_bf16(false, a.v, false, b.v,
                                                  (short)0, acc, false, false);
  }
  return acc;
}

// C/D fragment coords: VGPR i -> M = i + 8*(lane>=16), N = lane&15
__device__ __forceinline__ void tile_coords(int& row_off, int& col_off) {
  const int lane = threadIdx.x & 31;
  col_off = lane & 15;
  row_off = (lane >> 4) * 8;
}

// ---------------- conversion / packing ----------------
__global__ void cvt_pad_kernel(const float* __restrict__ src, unsigned short* __restrict__ dst,
                               int rows, int cols, int ldd) {
  int id = blockIdx.x * blockDim.x + threadIdx.x;
  if (id >= rows * ldd) return;
  int r = id / ldd, c = id - r * ldd;
  dst[id] = (c < cols) ? f2bf(src[(size_t)r * cols + c]) : (unsigned short)0;
}

__global__ void cvt_wembT_kernel(const float* __restrict__ wemb, unsigned short* __restrict__ dst) {
  int id = blockIdx.x * blockDim.x + threadIdx.x;   // over kNPAD*kD, dst[c,d]
  if (id >= kNPAD * kD) return;
  int c = id / kD, d = id - c * kD;
  dst[id] = (c < kC) ? f2bf(wemb[(size_t)d * kC + c]) : (unsigned short)0;
}

// ---------------- GEMM kernels (8 waves/block, one 16x16 tile per wave) -----
__global__ void gemm_embed_kernel(const unsigned short* __restrict__ xbf,
                                  const unsigned short* __restrict__ wbf,
                                  const float* __restrict__ b_emb,
                                  float* __restrict__ exf,
                                  unsigned short* __restrict__ exbf) {
  const int wave = threadIdx.x >> 5;
  const int m0 = blockIdx.x * 16;
  const int n0 = (blockIdx.y * 8 + wave) * 16;
  v8f acc = wmma_tile16(xbf, kKP, wbf, kKP, m0, n0, kKP);
  int ro, co; tile_coords(ro, co);
  const int col = n0 + co;
  const float freq = __expf((float)(col & ~1) * (-9.210340371976184f / (float)kD));
  const float bias = b_emb[col];
#pragma unroll
  for (int i = 0; i < 8; ++i) {
    int row = m0 + ro + i;
    int t = row & (kT - 1);
    float ang = (float)t * freq;
    float pe = (col & 1) ? cosf(ang) : sinf(ang);
    float v = acc[i] + bias + pe;
    exf [(size_t)row * kD + col] = v;
    exbf[(size_t)row * kD + col] = f2bf(v);
  }
}

__global__ void gemm_mlp1_kernel(const unsigned short* __restrict__ exbf,
                                 const unsigned short* __restrict__ w1,
                                 const float* __restrict__ tb1,
                                 unsigned short* __restrict__ hbf) {
  const int wave = threadIdx.x >> 5;
  const int m0 = blockIdx.x * 16;
  const int n0 = (blockIdx.y * 8 + wave) * 16;
  v8f acc = wmma_tile16(exbf, kD, w1, kD, m0, n0, kD);
  int ro, co; tile_coords(ro, co);
  const int col = n0 + co;
  const float b = tb1[col];
#pragma unroll
  for (int i = 0; i < 8; ++i) {
    float x = acc[i] + b;
    float g = 0.5f * x * (1.0f + erff(x * 0.7071067811865476f));   // exact GELU
    hbf[(size_t)(m0 + ro + i) * kD + col] = f2bf(g);
  }
}

__global__ void gemm_mlp2_kernel(const unsigned short* __restrict__ hbf,
                                 const unsigned short* __restrict__ w2,
                                 const float* __restrict__ tb2,
                                 const float* __restrict__ t_token,
                                 const float* __restrict__ tmask,
                                 float* __restrict__ out_t) {
  const int wave = threadIdx.x >> 5;
  const int m0 = blockIdx.x * 16;
  const int n0 = (blockIdx.y * 8 + wave) * 16;
  v8f acc = wmma_tile16(hbf, kD, w2, kD, m0, n0, kD);
  int ro, co; tile_coords(ro, co);
  const int col = n0 + co;
  const float b = tb2[col], tok = t_token[col];
#pragma unroll
  for (int i = 0; i < 8; ++i) {
    int row = m0 + ro + i;
    float s = 1.0f / (1.0f + __expf(-(acc[i] + b)));
    out_t[(size_t)row * kD + col] = (tmask[row] > 0.5f) ? tok : s;
  }
}

__global__ void gemm_proj_kernel(const unsigned short* __restrict__ mtdbf,
                                 const unsigned short* __restrict__ wT,
                                 float* __restrict__ mx) {
  const int wave = threadIdx.x >> 5;
  const int ntile = blockIdx.y * 8 + wave;
  if (ntile >= kNPAD / 16) return;                 // whole-wave skip, EXEC stays all-1
  const int m0 = blockIdx.x * 16;
  const int n0 = ntile * 16;
  v8f acc = wmma_tile16(mtdbf, kD, wT, kD, m0, n0, kD);
  int ro, co; tile_coords(ro, co);
  const int col = n0 + co;
  if (col >= kC) return;
#pragma unroll
  for (int i = 0; i < 8; ++i)
    mx[(size_t)(m0 + ro + i) * kC + col] = acc[i];
}

// ---------------- temporal score + top-k ----------------
__global__ void score_kernel(const float* __restrict__ exf, float* __restrict__ score) {
  __shared__ float s1[256], s2[256];
  const int m = blockIdx.x;
  const int t = m & (kT - 1);
  const int base = m - t;
  const int lo = (t > 23) ? (t - 23) : 0;
  const float denom = (float)((t + 1 < 24) ? (t + 1) : 24);
  float aM = 0.f, aV = 0.f;
  for (int d = threadIdx.x; d < kD; d += blockDim.x) {
    float sa = 0.f, sb = 0.f;
    for (int s = lo; s <= t; ++s) {
      float v = exf[(size_t)(base + s) * kD + d];
      sa += v; sb += v * v;
    }
    float m1 = sa / denom, m2 = sb / denom;
    aM += m1; aV += m2 - m1 * m1;
  }
  s1[threadIdx.x] = aM; s2[threadIdx.x] = aV;
  __syncthreads();
  for (int s = blockDim.x >> 1; s > 0; s >>= 1) {
    if (threadIdx.x < (unsigned)s) {
      s1[threadIdx.x] += s1[threadIdx.x + s];
      s2[threadIdx.x] += s2[threadIdx.x + s];
    }
    __syncthreads();
  }
  if (threadIdx.x == 0) score[m] = s2[0] / (s1[0] + 1e-6f);
}

// iterative-selection top-k, lax.top_k tie-break (lower index first)
__global__ void topk_kernel(const float* __restrict__ score, int n, int k,
                            float* __restrict__ idx_out, float* __restrict__ mask) {
  __shared__ float vals[512];
  __shared__ float sv[256];
  __shared__ int   si[256];
  const int b = blockIdx.x;
  const float* sc = score + (size_t)b * n;
  float* mk = mask + (size_t)b * n;
  for (int i = threadIdx.x; i < n; i += blockDim.x) { vals[i] = sc[i]; mk[i] = 0.f; }
  __syncthreads();
  for (int kk = 0; kk < k; ++kk) {
    float best = -FLT_MAX; int bi = n;
    for (int i = threadIdx.x; i < n; i += blockDim.x) {
      float v = vals[i];
      if (v > best) { best = v; bi = i; }
    }
    sv[threadIdx.x] = best; si[threadIdx.x] = bi;
    __syncthreads();
    for (int s = blockDim.x >> 1; s > 0; s >>= 1) {
      if (threadIdx.x < (unsigned)s) {
        float ov = sv[threadIdx.x + s]; int oi = si[threadIdx.x + s];
        if (ov > sv[threadIdx.x] || (ov == sv[threadIdx.x] && oi < si[threadIdx.x])) {
          sv[threadIdx.x] = ov; si[threadIdx.x] = oi;
        }
      }
      __syncthreads();
    }
    if (threadIdx.x == 0) {
      int w = si[0];
      idx_out[(size_t)b * k + kk] = (float)w;
      mk[w] = 1.f;
      vals[w] = -FLT_MAX;
    }
    __syncthreads();
  }
}

// ---------------- direct DFT / iDFT over T=512 ----------------
__global__ void dft_kernel(const float* __restrict__ exf,
                           float* __restrict__ cre, float* __restrict__ cim) {
  __shared__ float col[kT];
  const int bd = blockIdx.x;
  const int b = bd >> 9, d = bd & (kD - 1);
  for (int t = threadIdx.x; t < kT; t += blockDim.x)
    col[t] = exf[(size_t)(b * kT + t) * kD + d];
  __syncthreads();
  const float w = 2.f * kPI / (float)kT;
  for (int f = threadIdx.x; f < kF; f += blockDim.x) {
    float re = 0.f, im = 0.f;
    for (int t = 0; t < kT; ++t) {
      int ph = (f * t) & (kT - 1);                // exact phase reduction
      float s, c; __sincosf(w * (float)ph, &s, &c);
      re += col[t] * c;
      im -= col[t] * s;
    }
    size_t o = (size_t)bd * kF + f;
    cre[o] = re; cim[o] = im;
  }
}

__global__ void magmean_kernel(const float* __restrict__ cre, const float* __restrict__ cim,
                               float* __restrict__ mm) {
  int id = blockIdx.x * blockDim.x + threadIdx.x;
  if (id >= kB * kF) return;
  int b = id / kF, f = id - b * kF;
  float s = 0.f;
  for (int d = 0; d < kD; ++d) {
    size_t o = (size_t)(b * kD + d) * kF + f;
    s += sqrtf(cre[o] * cre[o] + cim[o] * cim[o]);
  }
  mm[id] = s / (float)kD;
}

__global__ void tm_kernel(const float* __restrict__ fmask, float* __restrict__ tm) {
  int id = blockIdx.x * blockDim.x + threadIdx.x;
  if (id >= kM) return;
  int b = id >> 9, t = id & (kT - 1);
  const float* fm = fmask + (size_t)b * kF;
  const float w = 2.f * kPI / (float)kT;
  float y = fm[0] + fm[kF - 1] * ((t & 1) ? -1.f : 1.f);
  for (int f = 1; f < kF - 1; ++f) {
    int ph = (f * t) & (kT - 1);
    y += 2.f * fm[f] * __cosf(w * (float)ph);
  }
  y *= (1.f / (float)kT);
  tm[id] = (y != 0.f) ? 1.f : 0.f;
}

__global__ void irfft_kernel(const float* __restrict__ cre, const float* __restrict__ cim,
                             const float* __restrict__ fmask,
                             const float* __restrict__ tokre, const float* __restrict__ tokim,
                             unsigned short* __restrict__ mtdbf) {
  __shared__ float re[kF], im[kF];
  const int bd = blockIdx.x;
  const int b = bd >> 9, d = bd & (kD - 1);
  const float* fm = fmask + (size_t)b * kF;
  const float tre = tokre[d], tim = tokim[d];
  for (int f = threadIdx.x; f < kF; f += blockDim.x) {
    size_t o = (size_t)bd * kF + f;
    bool msk = fm[f] > 0.5f;
    re[f] = msk ? tre : cre[o];
    im[f] = msk ? tim : cim[o];
  }
  __syncthreads();
  const float w = 2.f * kPI / (float)kT;
  for (int t = threadIdx.x; t < kT; t += blockDim.x) {
    float acc = re[0] + re[kF - 1] * ((t & 1) ? -1.f : 1.f);  // irfft ignores DC/Nyq imag
    for (int f = 1; f < kF - 1; ++f) {
      int ph = (f * t) & (kT - 1);
      float s, c; __sincosf(w * (float)ph, &s, &c);
      acc += 2.f * (re[f] * c - im[f] * s);
    }
    acc *= (1.f / (float)kT);
    mtdbf[(size_t)(b * kT + t) * kD + d] = f2bf(acc);
  }
}

// ---------------- per-element scalar MLP, one wave32 per element ------------
__global__ void fmlp_kernel(const float* __restrict__ mx, const float* __restrict__ tm,
                            const float* __restrict__ fW1, const float* __restrict__ fb1,
                            const float* __restrict__ fW2, const float* __restrict__ fb2,
                            float* __restrict__ out_f) {
  __shared__ float sw1[kD], sb1[kD], sw2[kD];
  for (int d = threadIdx.x; d < kD; d += blockDim.x) {
    sw1[d] = fW1[d]; sb1[d] = fb1[d]; sw2[d] = fW2[d];
  }
  __syncthreads();
  const int wave = threadIdx.x >> 5, lane = threadIdx.x & 31;
  const int e = blockIdx.x * (blockDim.x >> 5) + wave;
  if (e >= kM * kC) return;
  const float v = mx[e];
  float p = 0.f;
  for (int d = lane; d < kD; d += 32) {
    float u = fmaf(v, sw1[d], sb1[d]);
    float g = 0.5f * u * (1.0f + erff(u * 0.7071067811865476f));
    p = fmaf(sw2[d], g, p);
  }
#pragma unroll
  for (int off = 16; off > 0; off >>= 1) p += __shfl_xor(p, off);
  if (lane == 0) {
    float of = 1.0f / (1.0f + __expf(-(p + fb2[0])));
    out_f[e] = (tm[e / kC] != 0.f) ? v : of;
  }
}

} // anonymous namespace

extern "C" void kernel_launch(void* const* d_in, const int* in_sizes, int n_in,
                              void* d_out, int out_size, void* d_ws, size_t ws_size,
                              hipStream_t stream) {
  (void)in_sizes; (void)n_in; (void)out_size; (void)ws_size;
  const float* x      = (const float*)d_in[0];
  const float* W_emb  = (const float*)d_in[1];
  const float* b_emb  = (const float*)d_in[2];
  const float* t_tok  = (const float*)d_in[3];
  const float* tW1    = (const float*)d_in[4];
  const float* tb1    = (const float*)d_in[5];
  const float* tW2    = (const float*)d_in[6];
  const float* tb2    = (const float*)d_in[7];
  const float* ftokre = (const float*)d_in[8];
  const float* ftokim = (const float*)d_in[9];
  const float* fW1    = (const float*)d_in[10];
  const float* fb1    = (const float*)d_in[11];
  const float* fW2    = (const float*)d_in[12];
  const float* fb2    = (const float*)d_in[13];
  float* out = (float*)d_out;

  char* ws = (char*)d_ws;
  size_t off = 0;
  auto take = [&](size_t bytes) -> char* {
    char* p = ws + off;
    off = (off + bytes + 255) & ~(size_t)255;
    return p;
  };
  unsigned short* x_bf   = (unsigned short*)take((size_t)kM * kKP * 2);
  unsigned short* w_bf   = (unsigned short*)take((size_t)kD * kKP * 2);
  unsigned short* w1_bf  = (unsigned short*)take((size_t)kD * kD * 2);
  unsigned short* w2_bf  = (unsigned short*)take((size_t)kD * kD * 2);
  unsigned short* wt_bf  = (unsigned short*)take((size_t)kNPAD * kD * 2);
  float*          exf    = (float*)take((size_t)kM * kD * 4);
  unsigned short* ex_bf  = (unsigned short*)take((size_t)kM * kD * 2);
  unsigned short* h_bf   = (unsigned short*)take((size_t)kM * kD * 2);
  float*          scoreb = (float*)take((size_t)kM * 4);
  float*          tmaskf = (float*)take((size_t)kM * 4);
  float*          cre    = (float*)take((size_t)kB * kD * kF * 4);
  float*          cim    = (float*)take((size_t)kB * kD * kF * 4);
  float*          mmb    = (float*)take((size_t)kB * kF * 4);
  float*          fmaskf = (float*)take((size_t)kB * kF * 4);
  unsigned short* mtd_bf = (unsigned short*)take((size_t)kM * kD * 2);
  float*          mxb    = (float*)take((size_t)kM * kC * 4);
  float*          tmb    = (float*)take((size_t)kM * 4);

  // 0) pack / convert weights + activations to padded bf16
  { int n = kM * kKP;   cvt_pad_kernel<<<(n + 255) / 256, 256, 0, stream>>>(x,     x_bf,  kM, kC, kKP); }
  { int n = kD * kKP;   cvt_pad_kernel<<<(n + 255) / 256, 256, 0, stream>>>(W_emb, w_bf,  kD, kC, kKP); }
  { int n = kD * kD;    cvt_pad_kernel<<<(n + 255) / 256, 256, 0, stream>>>(tW1,   w1_bf, kD, kD, kD);  }
  { int n = kD * kD;    cvt_pad_kernel<<<(n + 255) / 256, 256, 0, stream>>>(tW2,   w2_bf, kD, kD, kD);  }
  { int n = kNPAD * kD; cvt_wembT_kernel<<<(n + 255) / 256, 256, 0, stream>>>(W_emb, wt_bf); }

  // 1) embedding GEMM + bias + positional encoding
  gemm_embed_kernel<<<dim3(kM / 16, kD / 16 / 8), 256, 0, stream>>>(x_bf, w_bf, b_emb, exf, ex_bf);

  // 2) temporal: windowed variance score -> top-k -> MLP -> out_t
  score_kernel<<<kM, 256, 0, stream>>>(exf, scoreb);
  topk_kernel<<<kB, 256, 0, stream>>>(scoreb, kT, kNMT, out + kOutTI, tmaskf);
  gemm_mlp1_kernel<<<dim3(kM / 16, kD / 16 / 8), 256, 0, stream>>>(ex_bf, w1_bf, tb1, h_bf);
  gemm_mlp2_kernel<<<dim3(kM / 16, kD / 16 / 8), 256, 0, stream>>>(h_bf, w2_bf, tb2, t_tok, tmaskf, out /*out_t @ 0*/);

  // 3) frequency: DFT -> magnitude top-k -> masked iDFT -> projection -> scalar MLP -> out_f
  dft_kernel<<<kB * kD, 256, 0, stream>>>(exf, cre, cim);
  magmean_kernel<<<(kB * kF + 255) / 256, 256, 0, stream>>>(cre, cim, mmb);
  topk_kernel<<<kB, 256, 0, stream>>>(mmb, kF, kNMF, out + kOutFI, fmaskf);
  tm_kernel<<<(kM + 255) / 256, 256, 0, stream>>>(fmaskf, tmb);
  irfft_kernel<<<kB * kD, 256, 0, stream>>>(cre, cim, fmaskf, ftokre, ftokim, mtd_bf);
  gemm_proj_kernel<<<dim3(kM / 16, 3), 256, 0, stream>>>(mtd_bf, wt_bf, mxb);
  fmlp_kernel<<<(kM * kC + 7) / 8, 256, 0, stream>>>(mxb, tmb, fW1, fb1, fW2, fb2, out + kOutF);
}
